// Head_5016521802174
// MI455X (gfx1250) — matrix operand order repaired and checked by reading, not compiled, not run on hardware
//
#include <hip/hip_runtime.h>

// MI455X / gfx1250, wave32. Flash-attention head.
//   K0: weight f32 -> f16 pre-convert (once).
//   K1: Q/K/V projection; LDS-staged A tile (converted once, bank-conflict-
//       free padding), f16 weights, WMMA f32_16x16x32_f16. V is written
//       TRANSPOSED (Vt[b][dk][S]) via a per-wave LDS bounce so the attention
//       kernel's P*V operand loads are contiguous.
//   K2: streaming softmax(QK^T/sqrt(dk))V, causal + length masks,
//       double-buffered async global->LDS staging (global_load_async_to_lds_b128
//       + s_wait_asynccnt), padded LDS rows (no bank conflicts).

typedef __attribute__((ext_vector_type(16))) _Float16 v16h;
typedef __attribute__((ext_vector_type(8)))  _Float16 v8h;
typedef __attribute__((ext_vector_type(8)))  float    v8f;
typedef __attribute__((ext_vector_type(4)))  int      v4i;

constexpr int Bn  = 8;
constexpr int Sn  = 2048;
constexpr int Dm  = 1024;
constexpr int DKn = 128;

constexpr int KP = 136;   // padded row stride (f16) for 128-wide rows: 272 B = 17x16 B
constexpr int VP = 40;    // padded row stride (f16) for 32-wide rows:   80 B = 5x16 B
constexpr int PP = 40;    // padded row stride (f16) for P rows (32 wide)

// ---- async global->LDS copy (CDNA5) --------------------------------------
#if __has_builtin(__builtin_amdgcn_global_load_async_to_lds_b128)
#define HAVE_ASYNC 1
typedef __attribute__((address_space(1))) v4i as1_v4i;
typedef __attribute__((address_space(3))) v4i as3_v4i;
__device__ __forceinline__ void async_copy16(const void* g, void* l)
{
    as1_v4i* gp = reinterpret_cast<as1_v4i*>(reinterpret_cast<uintptr_t>(g));
    as3_v4i* lp = reinterpret_cast<as3_v4i*>(
        static_cast<unsigned int>(reinterpret_cast<uintptr_t>(l)));
    __builtin_amdgcn_global_load_async_to_lds_b128(gp, lp, 0, 0);
}
#if __has_builtin(__builtin_amdgcn_s_wait_asynccnt)
#define WAIT_ASYNC(n) __builtin_amdgcn_s_wait_asynccnt(n)
#else
#define WAIT_ASYNC(n) asm volatile("s_wait_asynccnt %0" :: "i"(n) : "memory")
#endif
#else
#define HAVE_ASYNC 0
#endif

// ---------------------------------------------------------------------------
// Kernel 0: convert the three [DK, D] f32 weight matrices to f16 once.
// ---------------------------------------------------------------------------
__global__ __launch_bounds__(256)
void wconv_kernel(const float* __restrict__ Wq, const float* __restrict__ Wk,
                  const float* __restrict__ Wv, _Float16* __restrict__ Wh)
{
    const size_t per = (size_t)DKn * Dm;                     // 131072
    const size_t idx = ((size_t)blockIdx.x * 256 + threadIdx.x) * 8;
    const float* src = (idx < per) ? Wq : (idx < 2 * per) ? Wk : Wv;
    const float4* g  = (const float4*)(src + (idx % per));
    const float4 f0 = g[0], f1 = g[1];
    v8h h;
    h[0] = (_Float16)f0.x; h[1] = (_Float16)f0.y; h[2] = (_Float16)f0.z; h[3] = (_Float16)f0.w;
    h[4] = (_Float16)f1.x; h[5] = (_Float16)f1.y; h[6] = (_Float16)f1.z; h[7] = (_Float16)f1.w;
    *(v8h*)(Wh + idx) = h;
}

// ---------------------------------------------------------------------------
// Kernel 1: QKV projection. blockIdx = (m-tile, batch, which-proj).
// 256 threads = 8 waves; wave w owns output columns [16w, 16w+16).
// which==0/1 -> Q/K row-major [S][DK]; which==2 -> Vt transposed [DK][S].
// ---------------------------------------------------------------------------
__global__ __launch_bounds__(256)
void qkv_proj_kernel(const float* __restrict__ x, const float* __restrict__ ctx,
                     const _Float16* __restrict__ Wh, const int* __restrict__ lengths,
                     _Float16* __restrict__ Qo, _Float16* __restrict__ Ko,
                     _Float16* __restrict__ Vt)
{
    __shared__ _Float16 ldsA[16 * KP];      // 4.25 KB, padded rows (no bank conflicts)

    const int lane  = threadIdx.x & 31;
    const int wave  = threadIdx.x >> 5;
    const int mt    = blockIdx.x;
    const int b     = blockIdx.y;
    const int which = blockIdx.z;

    const float*    in  = (which == 2) ? x : ctx;
    const _Float16* W   = Wh + (size_t)which * DKn * Dm;

    const int m0   = mt * 16;
    const int n0   = wave * 16;
    const int half = lane >> 4;
    const int l15  = lane & 15;
    const int ka   = half * 16;      // A k split {0-15}/{16-31}
    const int kb   = half * 16;      // B k split {0-15}/{16-31}

    // staging: thread t converts 8 f32 -> 8 f16 of row (t>>4), cols (t&15)*8
    const int srowA = threadIdx.x >> 4;
    const int scolA = (threadIdx.x & 15) * 8;
    const float*    agrow = in + ((size_t)b * Sn + m0 + srowA) * Dm;
    const _Float16* brow  = W + (size_t)(n0 + l15) * Dm;

    v8f c = {};
    for (int ks = 0; ks < Dm; ks += 128) {
        {   // stage + convert A chunk (each element converted exactly once)
            const float4* g = (const float4*)(agrow + ks + scolA);
            const float4 f0 = g[0], f1 = g[1];
            v8h h;
            h[0] = (_Float16)f0.x; h[1] = (_Float16)f0.y; h[2] = (_Float16)f0.z; h[3] = (_Float16)f0.w;
            h[4] = (_Float16)f1.x; h[5] = (_Float16)f1.y; h[6] = (_Float16)f1.z; h[7] = (_Float16)f1.w;
            *(v8h*)(ldsA + srowA * KP + scolA) = h;
        }
        __syncthreads();
#pragma unroll
        for (int kc = 0; kc < 4; ++kc) {
            v16h a, bm;
            const _Float16* ap = ldsA + l15 * KP + kc * 32 + ka;
#pragma unroll
            for (int i = 0; i < 8; ++i) { a[i] = ap[i]; a[i + 8] = ap[i + 8]; }
            const _Float16* bp = brow + ks + kc * 32 + kb;
#pragma unroll
            for (int i = 0; i < 16; ++i) bm[i] = bp[i];
            c = __builtin_amdgcn_wmma_f32_16x16x32_f16(false, a, false, bm,
                                                       (short)0, c, false, false);
        }
        __syncthreads();
    }

    const int len = lengths[b];
    if (which != 2) {
        _Float16* out = (which == 0) ? Qo : Ko;
#pragma unroll
        for (int r = 0; r < 8; ++r) {
            const int row = m0 + half * 8 + r;               // C layout
            const float v = (row < len) ? c[r] : 0.0f;       // outputMask
            out[((size_t)b * Sn + row) * DKn + n0 + l15] = (_Float16)v;
        }
    } else {
        // transpose the 16x16 C tile through a per-wave LDS region (reuse ldsA:
        // each wave owns 272 f16 = 16x17), then store contiguous Vt rows.
        _Float16* tb = ldsA + wave * 272;
#pragma unroll
        for (int r = 0; r < 8; ++r) {
            const int row = m0 + half * 8 + r;
            const float v = (row < len) ? c[r] : 0.0f;
            tb[(half * 8 + r) * 17 + l15] = (_Float16)v;     // [m][n], pad 17
        }
        // lane (l15, half) emits channel n0+l15, seq chunk m0 + half*8 .. +8
        v8h cv;
#pragma unroll
        for (int m = 0; m < 8; ++m) cv[m] = tb[(half * 8 + m) * 17 + l15];
        *(v8h*)(Vt + ((size_t)b * DKn + n0 + l15) * Sn + m0 + half * 8) = cv;
    }
}

// ---------------------------------------------------------------------------
// Kernel 2: flash attention. blockIdx = (q-block of 64 rows, batch).
// 4 waves; wave w owns q-tile [q0blk+16w, +16). Double-buffered async
// staging of K (row-major) and Vt (dk-major) tiles; online softmax; 16x128
// f32 accumulator in VGPRs. All LDS rows padded -> conflict-free b128 reads.
// ---------------------------------------------------------------------------
__global__ __launch_bounds__(128)
void flash_attn_kernel(const _Float16* __restrict__ Q, const _Float16* __restrict__ K,
                       const _Float16* __restrict__ Vt, const int* __restrict__ lengths,
                       const int* __restrict__ dec_p, float* __restrict__ out)
{
    __shared__ _Float16 ldsK[2][32 * KP];     // 2 x 8.5 KB  (32 key rows x 128 dk)
    __shared__ _Float16 ldsVt[2][128 * VP];   // 2 x 10 KB   (128 dk rows x 32 keys)
    __shared__ _Float16 ldsP[4][16 * PP];     // 5 KB per-wave P relayout

    const int lane  = threadIdx.x & 31;
    const int wave  = threadIdx.x >> 5;
    const int b     = blockIdx.y;
    const int q0blk = blockIdx.x * 64;
    const int q0    = q0blk + wave * 16;
    const int len   = lengths[b];
    const int dec   = dec_p[0];

    // log2 domain: scale = (1/sqrt(dk)) * log2(e)
    const float scale = 0.08838834764831845f * 1.44269504088896341f;

    const int half = lane >> 4;
    const int l15  = lane & 15;
    const int ka   = half * 16;
    const int kb   = half * 16;

    // ---- Q A-fragments (16 rows x 128) ----
    v16h qa[4];
    const _Float16* qrow = Q + ((size_t)b * Sn + q0 + l15) * DKn;
#pragma unroll
    for (int kc = 0; kc < 4; ++kc) {
        const _Float16* qp = qrow + kc * 32 + ka;
#pragma unroll
        for (int i = 0; i < 8; ++i) { qa[kc][i] = qp[i]; qa[kc][i + 8] = qp[i + 8]; }
    }

    v8f acc[8];
#pragma unroll
    for (int nt = 0; nt < 8; ++nt) acc[nt] = (v8f){};
    float rmax[8], rsum[8];
#pragma unroll
    for (int r = 0; r < 8; ++r) { rmax[r] = -__builtin_inff(); rsum[r] = 0.0f; }

    int jend_blk = len;
    if (dec) jend_blk = (jend_blk < q0blk + 64) ? jend_blk : (q0blk + 64);
    int jend_w = len;
    if (dec) jend_w = (jend_w < q0 + 16) ? jend_w : (q0 + 16);

    const int srowK = threadIdx.x >> 2;         // 4 threads per key row (K tile)
    const int ssegK = (threadIdx.x & 3) * 32;   // 32 f16 (64 B) per thread
    const int chV   = threadIdx.x;              // 1 thread per dk channel (Vt tile)

    auto stage = [&](int buf, int j0) {
        const _Float16* gK = K  + ((size_t)b * Sn + j0 + srowK) * DKn + ssegK;
        const _Float16* gV = Vt + ((size_t)b * DKn + chV) * Sn + j0;
        _Float16* lK = &ldsK[buf][srowK * KP + ssegK];
        _Float16* lV = &ldsVt[buf][chV * VP];
#if HAVE_ASYNC
#pragma unroll
        for (int i = 0; i < 4; ++i) {
            async_copy16(gK + i * 8, lK + i * 8);
            async_copy16(gV + i * 8, lV + i * 8);
        }
#else
#pragma unroll
        for (int i = 0; i < 4; ++i) {
            *(v8h*)(lK + i * 8) = *(const v8h*)(gK + i * 8);
            *(v8h*)(lV + i * 8) = *(const v8h*)(gV + i * 8);
        }
#endif
    };

    const int nIter = (jend_blk + 31) >> 5;
    stage(0, 0);                                 // prime buffer 0

    for (int it = 0; it < nIter; ++it) {
        const int  j0  = it << 5;
        const int  buf = it & 1;
        const bool pf  = (it + 1 < nIter);
        if (pf) stage(buf ^ 1, j0 + 32);         // prefetch next tile
#if HAVE_ASYNC
        if (pf) { WAIT_ASYNC(8); } else { WAIT_ASYNC(0); }   // current tile landed
#endif
        __syncthreads();

        if (j0 < jend_w) {   // wave-uniform: EXEC all-ones around WMMA
            const _Float16* lK = &ldsK[buf][0];
            const _Float16* lV = &ldsVt[buf][0];

            // ---- S = Q * K^T, two 16-key subtiles ----
            v8f s[2];
#pragma unroll
            for (int t = 0; t < 2; ++t) {
                s[t] = (v8f){};
#pragma unroll
                for (int kc = 0; kc < 4; ++kc) {
                    v16h kf;
                    const _Float16* kp = lK + (t * 16 + l15) * KP + kc * 32 + kb;
#pragma unroll
                    for (int i = 0; i < 16; ++i) kf[i] = kp[i];
                    s[t] = __builtin_amdgcn_wmma_f32_16x16x32_f16(false, qa[kc], false, kf,
                                                                  (short)0, s[t], false, false);
                }
            }

            // ---- mask + online softmax over 32 keys ----
            float p[2][8], f[8];
#pragma unroll
            for (int r = 0; r < 8; ++r) {
                const int qi = q0 + half * 8 + r;
                float v = -__builtin_inff();
#pragma unroll
                for (int t = 0; t < 2; ++t) {
                    const int key = j0 + t * 16 + l15;
                    const bool ok = (key < len) && (!dec || key <= qi);
                    p[t][r] = ok ? s[t][r] * scale : -__builtin_inff();
                    v = fmaxf(v, p[t][r]);
                }
                for (int m = 1; m <= 8; m <<= 1) v = fmaxf(v, __shfl_xor(v, m, 32));
                const float nm = fmaxf(rmax[r], v);
                float ps = 0.0f;
#pragma unroll
                for (int t = 0; t < 2; ++t) { p[t][r] = exp2f(p[t][r] - nm); ps += p[t][r]; }
                for (int m = 1; m <= 8; m <<= 1) ps += __shfl_xor(ps, m, 32);
                f[r]    = exp2f(rmax[r] - nm);
                rsum[r] = rsum[r] * f[r] + ps;
                rmax[r] = nm;
            }
#pragma unroll
            for (int nt = 0; nt < 8; ++nt)
#pragma unroll
                for (int r = 0; r < 8; ++r) acc[nt][r] *= f[r];

            // ---- relayout P (C-layout) -> A-operand via per-wave LDS buffer ----
            _Float16* pbuf = &ldsP[wave][0];
#pragma unroll
            for (int t = 0; t < 2; ++t)
#pragma unroll
                for (int r = 0; r < 8; ++r)
                    pbuf[(half * 8 + r) * PP + t * 16 + l15] = (_Float16)p[t][r];
            v16h pa;
            {
                const _Float16* pp = pbuf + l15 * PP + ka;   // DS in-order within wave
#pragma unroll
                for (int i = 0; i < 8; ++i) { pa[i] = pp[i]; pa[i + 8] = pp[i + 8]; }
            }

            // ---- acc += P * V : B fragment is contiguous 32 B per lane ----
#pragma unroll
            for (int nt = 0; nt < 8; ++nt) {
                v16h vf;
                const _Float16* vp = lV + (nt * 16 + l15) * VP + kb;
#pragma unroll
                for (int i = 0; i < 16; ++i) vf[i] = vp[i];
                acc[nt] = __builtin_amdgcn_wmma_f32_16x16x32_f16(false, pa, false, vf,
                                                                 (short)0, acc[nt], false, false);
            }
        }
        __syncthreads();
    }

    // ---- normalize and store f32 output ----
#pragma unroll
    for (int nt = 0; nt < 8; ++nt)
#pragma unroll
        for (int r = 0; r < 8; ++r) {
            const int row = q0 + half * 8 + r;
            out[((size_t)b * Sn + row) * DKn + nt * 16 + l15] = acc[nt][r] / rsum[r];
        }
}

// ---------------------------------------------------------------------------
extern "C" void kernel_launch(void* const* d_in, const int* in_sizes, int n_in,
                              void* d_out, int out_size, void* d_ws, size_t ws_size,
                              hipStream_t stream)
{
    const float* x       = (const float*)d_in[0];
    const float* ctx     = (const float*)d_in[1];
    const float* Wq      = (const float*)d_in[2];
    const float* Wk      = (const float*)d_in[3];
    const float* Wv      = (const float*)d_in[4];
    const int*   lengths = (const int*)d_in[5];
    const int*   dec     = (const int*)d_in[7];
    float*       out     = (float*)d_out;

    // workspace: Q,K row-major; Vt transposed [B][DK][S]; f16 weights
    _Float16* Qw  = (_Float16*)d_ws;
    _Float16* Kw  = Qw + (size_t)Bn * Sn * DKn;
    _Float16* Vtw = Kw + (size_t)Bn * Sn * DKn;
    _Float16* Wh  = Vtw + (size_t)Bn * Sn * DKn;

    wconv_kernel<<<dim3(192), 256, 0, stream>>>(Wq, Wk, Wv, Wh);

    dim3 g1(Sn / 16, Bn, 3);
    qkv_proj_kernel<<<g1, 256, 0, stream>>>(x, ctx, Wh, lengths, Qw, Kw, Vtw);

    dim3 g2(Sn / 64, Bn);
    flash_attn_kernel<<<g2, 128, 0, stream>>>(Qw, Kw, Vtw, lengths, dec, out);
}
